// Qatten_88081189307114
// MI455X (gfx1250) — compile-verified, matrix operand order checked by reading, not run on hardware
//
#include <hip/hip_runtime.h>
#include <cstdint>

// Problem dims
#define NA 8
#define BB 32768
#define DD 128
#define AA 64
#define EE 32
#define HH 128

#define BT 128
#define THREADS 256

// Workspace layout (bf16 element offsets): pre-transposed bf16 weights [out][K]
#define OFF_W1T  0                        // N*H*D = 131072
#define OFF_W2T  (OFF_W1T + NA*HH*DD)     // 131072
#define OFF_W3T  (OFF_W2T + NA*HH*HH)     // 262144  (N*A*H = 65536)
#define OFF_WC1T (OFF_W3T + NA*AA*HH)     // 327680  (E*D = 4096)
#define W_TOTAL  (OFF_WC1T + EE*DD)       // 331776 ushorts = 663552 B

// Dynamic LDS layout (bytes)
#define SM_X   0        // 128x128 bf16 = 32768
#define SM_WA  32768    // weight buffer A (W1t, later W3t)   32768
#define SM_WB  65536    // weight buffer B (W2t)              32768
#define SM_H   98304    // 128x128 bf16                       32768
#define SM_WC  131072   // 32x128 bf16 = 8192
#define SM_WS1 139264   // 128 f32
#define SM_G   139776   // 128 f32
#define SM_C   140288   // 128 f32
#define SMEM_BYTES 140800

#if __has_builtin(__builtin_amdgcn_global_load_async_to_lds_b128)
#  define HAVE_ASYNC 1
#else
#  define HAVE_ASYNC 0
#endif

typedef __attribute__((ext_vector_type(16))) __bf16 bf16x16;
typedef __attribute__((ext_vector_type(8)))  float  f32x8;
typedef __attribute__((ext_vector_type(4)))  int    v4i;

union FragU { uint4 u[2]; bf16x16 v; };

__device__ __forceinline__ unsigned short f2bf(float f) {
  unsigned int u = __float_as_uint(f);
  unsigned int r = 0x7FFFu + ((u >> 16) & 1u);   // round-to-nearest-even
  return (unsigned short)((u + r) >> 16);
}
__device__ __forceinline__ float bf2f(unsigned int h16) {
  return __uint_as_float((h16 & 0xFFFFu) << 16);
}

#if HAVE_ASYNC
typedef __attribute__((address_space(1))) v4i GV4;  // global int4
typedef __attribute__((address_space(3))) v4i LV4;  // LDS int4
__device__ __forceinline__ void async_cp16(const unsigned short* g, unsigned short* l) {
  // per-lane 16B async DMA: LDS[l] = MEM[g]; tracked by ASYNCcnt, no VGPR data path
  __builtin_amdgcn_global_load_async_to_lds_b128(
      (GV4*)(uintptr_t)g, (LV4*)(unsigned int)(uintptr_t)l, 0, 0);
}
#endif

// Issue a (possibly async) LDS staging copy of `bytes` from workspace bf16.
__device__ __forceinline__ void stage_issue(const unsigned short* g, unsigned short* l,
                                            int bytes, int t) {
  const int chunks = bytes >> 4;
#if HAVE_ASYNC
  for (int i = t; i < chunks; i += THREADS) async_cp16(g + i * 8, l + i * 8);
#else
  for (int i = t; i < chunks; i += THREADS) ((uint4*)l)[i] = ((const uint4*)g)[i];
#endif
}
__device__ __forceinline__ void wait_async0() {
#if HAVE_ASYNC
#  if __has_builtin(__builtin_amdgcn_s_wait_asynccnt)
  __builtin_amdgcn_s_wait_asynccnt(0);
#  else
  asm volatile("s_wait_asynccnt 0" ::: "memory");
#  endif
#endif
}

// A/B fragment load for v_wmma_f32_16x16x32_bf16 from row-major bf16 [16 x K] tile.
// Per ISA 7.12.2: lanes 0-15 hold K={0..7,16..23}, lanes 16-31 hold K={8..15,24..31}.
// B operand uses the identical per-lane layout with lane -> row of Wt (= column of W).
__device__ __forceinline__ bf16x16 ld_frag(const unsigned short* p, int ld, int lane) {
  const int m  = lane & 15;
  const int hl = lane >> 4;
  FragU f;
  f.u[0] = *(const uint4*)(p + m * ld + 8 * hl);
  f.u[1] = *(const uint4*)(p + m * ld + 16 + 8 * hl);
  return f.v;
}

__device__ __forceinline__ f32x8 wmma_bf16(bf16x16 a, bf16x16 b, f32x8 c) {
  return __builtin_amdgcn_wmma_f32_16x16x32_bf16(false, a, false, b, (short)0, c, false, false);
}
__device__ __forceinline__ void vzero(f32x8& v) {
  for (int j = 0; j < 8; ++j) v[j] = 0.f;
}

// ---------------------------------------------------------------------------
// K0: convert + transpose weights to bf16 in workspace
// ---------------------------------------------------------------------------
__global__ void prep_weights(const float* __restrict__ W1, const float* __restrict__ W2,
                             const float* __restrict__ W3, const float* __restrict__ Wc1,
                             unsigned short* __restrict__ ws) {
  for (int i = blockIdx.x * blockDim.x + threadIdx.x; i < W_TOTAL;
       i += gridDim.x * blockDim.x) {
    float v;
    if (i < OFF_W2T) {                       // W1t[n][h][d] = W1[n][d][h]
      int r = i, n = r / (HH * DD), q = r % (HH * DD), h = q / DD, d = q % DD;
      v = W1[((size_t)n * DD + d) * HH + h];
    } else if (i < OFF_W3T) {                // W2t[n][h][k] = W2[n][k][h]
      int r = i - OFF_W2T, n = r / (HH * HH), q = r % (HH * HH), h = q / HH, k = q % HH;
      v = W2[((size_t)n * HH + k) * HH + h];
    } else if (i < OFF_WC1T) {               // W3t[n][a][h] = W3[n][h][a]
      int r = i - OFF_W3T, n = r / (AA * HH), q = r % (AA * HH), a = q / HH, h = q % HH;
      v = W3[((size_t)n * HH + h) * AA + a];
    } else {                                 // Wc1t[e][d] = Wc1[d][e]
      int r = i - OFF_WC1T, e = r / DD, d = r % DD;
      v = Wc1[(size_t)d * EE + e];
    }
    ws[i] = f2bf(v);
  }
}

// ---------------------------------------------------------------------------
// Main fused kernel: per 128-row batch tile, loop over 8 agents.
// out[b,a] = (1/8) * sum_n sigmoid(s_n[b]) * (DQN_n(x_nb))[a] + (1/8) * sum_n c_n[b]
// (softmax row-sum == 1 identity eliminates the query/key/score path)
// ---------------------------------------------------------------------------
__global__ void __launch_bounds__(THREADS, 1) qatten_main(
    const float* __restrict__ states,
    const float* __restrict__ b1, const float* __restrict__ b2, const float* __restrict__ b3,
    const float* __restrict__ Ws1, const float* __restrict__ bs1,
    const float* __restrict__ Ws2, const float* __restrict__ bs2,
    const float* __restrict__ bc1, const float* __restrict__ Wc2, const float* __restrict__ bc2,
    const unsigned short* __restrict__ wsw, float* __restrict__ out) {
  extern __shared__ __align__(16) char smem[];
  unsigned short* sX  = (unsigned short*)(smem + SM_X);
  unsigned short* sWA = (unsigned short*)(smem + SM_WA);
  unsigned short* sWB = (unsigned short*)(smem + SM_WB);
  unsigned short* sH  = (unsigned short*)(smem + SM_H);
  unsigned short* sWc = (unsigned short*)(smem + SM_WC);
  float* sWs1 = (float*)(smem + SM_WS1);
  float* sG   = (float*)(smem + SM_G);
  float* sC   = (float*)(smem + SM_C);

  const int t    = threadIdx.x;
  const int lane = t & 31;
  const int w    = t >> 5;       // wave id 0..7 == M-tile
  const int hl   = lane >> 4;    // half of wave
  const int ln   = lane & 15;
  const int r0   = blockIdx.x * BT;

  if (t < DD) sWs1[t] = Ws1[t];
  if (t < BT) sC[t] = 0.f;
  const float bs1v = bs1[0], ws2v = Ws2[0], bs2v = bs2[0], bc2v = bc2[0];
  const float bc1v0 = bc1[ln], bc1v1 = bc1[16 + ln];
  const float wc2v0 = Wc2[ln], wc2v1 = Wc2[16 + ln];

  f32x8 oacc[4];
  for (int nt = 0; nt < 4; ++nt) vzero(oacc[nt]);

  for (int n = 0; n < NA; ++n) {
    __syncthreads();  // (a) all reads of prev-agent LDS contents complete

    // ---- stage X tile (contiguous 128x128 f32) -> bf16 LDS (needs conversion) ----
    const float* xg = states + ((size_t)n * BB + r0) * DD;
    for (int i = 0; i < 16; ++i) {
      int idx4 = t + i * THREADS;                    // float4 index, 0..4095
      float4 v = ((const float4*)xg)[idx4];
      unsigned int lo = (unsigned)f2bf(v.x) | ((unsigned)f2bf(v.y) << 16);
      unsigned int hi = (unsigned)f2bf(v.z) | ((unsigned)f2bf(v.w) << 16);
      ((uint2*)sX)[idx4] = make_uint2(lo, hi);
    }
    // ---- stage W1t + Wc1t (async DMA to LDS when available) ----
    stage_issue(wsw + OFF_W1T + n * HH * DD, sWA, HH * DD * 2, t);
    stage_issue(wsw + OFF_WC1T, sWc, EE * DD * 2, t);
    // ---- prefetch next agent's states tile into cache while we compute ----
    if (n + 1 < NA) {
      const char* nx = (const char*)(states + ((size_t)(n + 1) * BB + r0) * DD);
      __builtin_prefetch(nx + t * 256, 0, 3);
      __builtin_prefetch(nx + t * 256 + 128, 0, 3);
    }
    wait_async0();
    __syncthreads();  // (b) sX, sWA (W1t), sWc visible to all waves

    // issue W2t staging now; it lands while gate/cGEMM/GEMM1 run
    stage_issue(wsw + OFF_W2T + n * HH * HH, sWB, HH * HH * 2, t);

    // ---- gate: sigmoid(relu(x . Ws1 + bs1) * Ws2 + bs2) / 8 ----
    if (t < BT) {
      float acc = 0.f;
      const uint4* rp = (const uint4*)(sX + t * DD);
      for (int i = 0; i < 16; ++i) {         // 8 bf16 per uint4
        uint4 u = rp[i];
        const float* wp = sWs1 + i * 8;
        acc += bf2f(u.x) * wp[0] + bf2f(u.x >> 16) * wp[1];
        acc += bf2f(u.y) * wp[2] + bf2f(u.y >> 16) * wp[3];
        acc += bf2f(u.z) * wp[4] + bf2f(u.z >> 16) * wp[5];
        acc += bf2f(u.w) * wp[6] + bf2f(u.w >> 16) * wp[7];
      }
      float s = fmaxf(acc + bs1v, 0.f) * ws2v + bs2v;
      sG[t] = 0.125f / (1.f + __expf(-s));
    }

    // ---- constraint: relu(X @ Wc1 + bc1) . Wc2 + bc2, accumulated /8 ----
    {
      f32x8 hc0, hc1; vzero(hc0); vzero(hc1);
      const unsigned short* ax = sX + w * 16 * DD;
      for (int kk = 0; kk < 4; ++kk) {
        bf16x16 a   = ld_frag(ax + kk * 32, DD, lane);
        bf16x16 bf0 = ld_frag(sWc + 0 * 16 * DD + kk * 32, DD, lane);
        bf16x16 bf1 = ld_frag(sWc + 1 * 16 * DD + kk * 32, DD, lane);
        hc0 = wmma_bf16(a, bf0, hc0);
        hc1 = wmma_bf16(a, bf1, hc1);
      }
      for (int j = 0; j < 8; ++j) {
        float h0 = fmaxf(hc0[j] + bc1v0, 0.f);
        float h1 = fmaxf(hc1[j] + bc1v1, 0.f);
        float p = h0 * wc2v0 + h1 * wc2v1;
        p += __shfl_xor(p, 1, 16);
        p += __shfl_xor(p, 2, 16);
        p += __shfl_xor(p, 4, 16);
        p += __shfl_xor(p, 8, 16);
        if (ln == 0) sC[w * 16 + j + 8 * hl] += (p + bc2v) * 0.125f;
      }
    }

    // ---- GEMM1: H1 = relu(X @ W1 + b1) -> sH ----
    f32x8 acc[8];
    {
      for (int nt = 0; nt < 8; ++nt) vzero(acc[nt]);
      const unsigned short* ax = sX + w * 16 * DD;
      for (int kk = 0; kk < 4; ++kk) {
        bf16x16 a = ld_frag(ax + kk * 32, DD, lane);
        for (int nt = 0; nt < 8; ++nt) {
          bf16x16 bf = ld_frag(sWA + nt * 16 * DD + kk * 32, DD, lane);
          acc[nt] = wmma_bf16(a, bf, acc[nt]);
        }
      }
      for (int nt = 0; nt < 8; ++nt) {
        int col = nt * 16 + ln;
        float bb = b1[n * HH + col];
        for (int j = 0; j < 8; ++j) {
          int row = w * 16 + j + 8 * hl;
          sH[row * HH + col] = f2bf(fmaxf(acc[nt][j] + bb, 0.f));
        }
      }
    }
    wait_async0();
    __syncthreads();  // (c) sH visible; sWB (W2t) landed; all sWA reads done

    // issue W3t staging into sWA (safe: GEMM1 reads finished at (c))
    stage_issue(wsw + OFF_W3T + n * AA * HH, sWA, AA * HH * 2, t);

    // ---- GEMM2: H2 = relu(H1 @ W2 + b2) -> sX (reuse) ----
    {
      for (int nt = 0; nt < 8; ++nt) vzero(acc[nt]);
      const unsigned short* ah = sH + w * 16 * HH;
      for (int kk = 0; kk < 4; ++kk) {
        bf16x16 a = ld_frag(ah + kk * 32, HH, lane);
        for (int nt = 0; nt < 8; ++nt) {
          bf16x16 bf = ld_frag(sWB + nt * 16 * HH + kk * 32, HH, lane);
          acc[nt] = wmma_bf16(a, bf, acc[nt]);
        }
      }
      for (int nt = 0; nt < 8; ++nt) {
        int col = nt * 16 + ln;
        float bb = b2[n * HH + col];
        for (int j = 0; j < 8; ++j) {
          int row = w * 16 + j + 8 * hl;
          sX[row * HH + col] = f2bf(fmaxf(acc[nt][j] + bb, 0.f));
        }
      }
    }
    wait_async0();
    __syncthreads();  // (d) sX (=H2) visible; sWA (W3t) landed

    // ---- GEMM3: Q = H2 @ W3 + b3 ; gated accumulate into registers ----
    {
      f32x8 q[4];
      for (int nt = 0; nt < 4; ++nt) vzero(q[nt]);
      const unsigned short* ah = sX + w * 16 * HH;
      for (int kk = 0; kk < 4; ++kk) {
        bf16x16 a = ld_frag(ah + kk * 32, HH, lane);
        for (int nt = 0; nt < 4; ++nt) {
          bf16x16 bf = ld_frag(sWA + nt * 16 * HH + kk * 32, HH, lane);
          q[nt] = wmma_bf16(a, bf, q[nt]);
        }
      }
      float b3v[4];
      for (int nt = 0; nt < 4; ++nt) b3v[nt] = b3[n * AA + nt * 16 + ln];
      for (int j = 0; j < 8; ++j) {
        float g = sG[w * 16 + j + 8 * hl];
        for (int nt = 0; nt < 4; ++nt)
          oacc[nt][j] += g * (q[nt][j] + b3v[nt]);
      }
    }
  }

  __syncthreads();
  // ---- epilogue: out = gated-mean(Q) + mean(c) ----
  for (int nt = 0; nt < 4; ++nt) {
    int col = nt * 16 + ln;
    for (int j = 0; j < 8; ++j) {
      int row = w * 16 + j + 8 * hl;
      out[(size_t)(r0 + row) * AA + col] = oacc[nt][j] + sC[row];
    }
  }
}

extern "C" void kernel_launch(void* const* d_in, const int* in_sizes, int n_in,
                              void* d_out, int out_size, void* d_ws, size_t ws_size,
                              hipStream_t stream) {
  (void)in_sizes; (void)n_in; (void)out_size; (void)ws_size;
  const float* states = (const float*)d_in[0];
  const float* W1  = (const float*)d_in[1];
  const float* b1  = (const float*)d_in[2];
  const float* W2  = (const float*)d_in[3];
  const float* b2  = (const float*)d_in[4];
  const float* W3  = (const float*)d_in[5];
  const float* b3  = (const float*)d_in[6];
  // d_in[7..12] = Wq1,bq1,Wq2,bq2,Wk,bk : eliminated (softmax row-sum == 1)
  const float* Ws1 = (const float*)d_in[13];
  const float* bs1 = (const float*)d_in[14];
  const float* Ws2 = (const float*)d_in[15];
  const float* bs2 = (const float*)d_in[16];
  const float* Wc1 = (const float*)d_in[17];
  const float* bc1 = (const float*)d_in[18];
  const float* Wc2 = (const float*)d_in[19];
  const float* bc2 = (const float*)d_in[20];
  unsigned short* wsb = (unsigned short*)d_ws;
  float* out = (float*)d_out;

  prep_weights<<<1296, 256, 0, stream>>>(W1, W2, W3, Wc1, wsb);
  qatten_main<<<BB / BT, THREADS, SMEM_BYTES, stream>>>(
      states, b1, b2, b3, Ws1, bs1, Ws2, bs2, bc1, Wc2, bc2, wsb, out);
}
